// DecoderLayer_14723147890845
// MI455X (gfx1250) — matrix operand order repaired and checked by reading, not compiled, not run on hardware
//
#include <hip/hip_runtime.h>

// ---------------------------------------------------------------------------
// Transformer decoder layer for MI455X (gfx1250, wave32, WMMA).
// All GEMMs run through v_wmma_f32_16x16x32_bf16 (bf16 A/B, f32 accumulate).
// Weights are pre-transposed + converted to bf16 once per call into d_ws.
//
// Input order (jax tree-flatten, dict keys sorted):
//  0 X (S*D f32)            1 enc_output (S*D f32)
//  2 look_ahead_mask (S*S i32)  3 padding_mask (S i32)
//  4 ffn.b1  5 ffn.b2  6 ffn.w1 (D*FF)  7 ffn.w2 (FF*D)
//  8 ln1.b 9 ln1.g 10 ln2.b 11 ln2.g 12 ln3.b 13 ln3.g
// 14 mha1.bk 15 mha1.bout 16 mha1.bq 17 mha1.bv
// 18 mha1.wk 19 mha1.wout 20 mha1.wq 21 mha1.wv
// 22..29 mha2 in same order.
// Output: x3out (S*D) ++ aw1 (H*S*S) ++ aw2 (H*S*S), all f32.
// ---------------------------------------------------------------------------

#define S_LEN 2048
#define D_MOD 1024
#define N_HEADS 16
#define D_KH 64
#define D_FFN 4096

typedef __attribute__((ext_vector_type(16))) __bf16 v16bf;
typedef __attribute__((ext_vector_type(8)))  float  v8f;

__device__ __forceinline__ __bf16 to_bf(float f) {
  unsigned u = __builtin_bit_cast(unsigned, f);
  u += 0x7FFFu + ((u >> 16) & 1u);          // round-to-nearest-even
  unsigned short h = (unsigned short)(u >> 16);
  return __builtin_bit_cast(__bf16, h);
}
__device__ __forceinline__ __bf16 cvt_elem(float f)  { return to_bf(f); }
__device__ __forceinline__ __bf16 cvt_elem(__bf16 b) { return b; }

// A-fragment K index per VGPR v (16-bit A 16x32 layout, ISA 7.12.2):
// lanes 0-15 hold K 0..7 / 16..23 ; lanes 16-31 hold K 8..15 / 24..31.
__device__ __forceinline__ int a_kidx(int v, int hi) {
  int base = (v < 4) ? (2 * v) : (16 + 2 * (v - 4));
  return base + (hi ? 8 : 0);
}
// B-fragment K index per VGPR v (same K order as A: lanes 0-15 K 0..7 / 16..23 ; lanes 16-31 K 8..15 / 24..31)
__device__ __forceinline__ int b_kidx(int v, int hi) {
  return ((v < 4) ? (2 * v) : (16 + 2 * (v - 4))) + (hi ? 8 : 0);
}

template<typename AT>
__device__ __forceinline__ v16bf load_a_frag(const AT* __restrict__ A, int lda,
                                             int row0, int k0, int l, int hi) {
  v16bf a;
  const AT* p = A + (long)(row0 + l) * lda + k0;
#pragma unroll
  for (int v = 0; v < 8; ++v) {
    int k = a_kidx(v, hi);
    a[2 * v]     = cvt_elem(p[k]);
    a[2 * v + 1] = cvt_elem(p[k + 1]);
  }
  return a;
}

// B stored transposed: (N x K) row-major, ldb = K-stride.  B(k,n)=Bt[n*ldb+k]
__device__ __forceinline__ v16bf load_b_t(const __bf16* __restrict__ B, int ldb,
                                          int k0, int col0, int l, int hi) {
  v16bf b;
  const __bf16* p = B + (long)(col0 + l) * ldb + k0;
#pragma unroll
  for (int v = 0; v < 8; ++v) {
    int k = b_kidx(v, hi);
    b[2 * v]     = p[k];
    b[2 * v + 1] = p[k + 1];
  }
  return b;
}

// B stored normally: (K x N) row-major, ldb = N-stride.
__device__ __forceinline__ v16bf load_b_nt(const __bf16* __restrict__ B, int ldb,
                                           int k0, int col0, int l, int hi) {
  v16bf b;
  const __bf16* p = B + col0 + l;
#pragma unroll
  for (int v = 0; v < 8; ++v) {
    int k = k0 + b_kidx(v, hi);
    b[2 * v]     = p[(long)k * ldb];
    b[2 * v + 1] = p[(long)(k + 1) * ldb];
  }
  return b;
}

__device__ __forceinline__ v8f wmma_bf(v16bf a, v16bf b, v8f c) {
  return __builtin_amdgcn_wmma_f32_16x16x32_bf16(false, a, false, b,
                                                 (short)0, c, false, false);
}

// ---------------------------------------------------------------------------
// Generic batched WMMA GEMM.  C = act( scale*(A@B) + bias, mask )
// 256 threads = 8 waves as 4(M) x 2(N); wave tile 32 x (NT*16).
// Block tile 128(M) x (NT*32)(N).   M%128==0, N%(NT*32)==0, K%32==0 assumed.
// MM: 0 = no mask, 1 = causal (col>row -> -1e9), 2 = row-vector mask[col].
// ---------------------------------------------------------------------------
template<typename AT, bool BT, int NT, int MM>
__global__ void wmma_gemm(const AT* __restrict__ A, long batchA, int lda,
                          const __bf16* __restrict__ B, long batchB, int ldb,
                          const float* __restrict__ bias,
                          const int* __restrict__ rvmask,
                          float scale,
                          float* __restrict__ outF, long batchOF, int ldof,
                          __bf16* __restrict__ outB, long batchOB, int ldob,
                          int K, int relu) {
  const int wave = threadIdx.x >> 5;
  const int lane = threadIdx.x & 31;
  const int l = lane & 15, hi = lane >> 4;
  const int wm = wave >> 1, wn = wave & 1;
  const int bm = blockIdx.y * 128 + wm * 32;
  const int bn = blockIdx.x * (NT * 32) + wn * (NT * 16);
  const int z = blockIdx.z;

  const AT* Ab = A + (long)z * batchA;
  const __bf16* Bb = B + (long)z * batchB;

  v8f c[2][NT] = {};
  for (int k0 = 0; k0 < K; k0 += 32) {
    if (k0 + 64 < K) {  // prefetch next-next K tile (global_prefetch_b8)
      __builtin_prefetch(Ab + (long)(bm + l) * lda + k0 + 64, 0, 1);
      if constexpr (BT)
        __builtin_prefetch(Bb + (long)(bn + l) * ldb + k0 + 64, 0, 1);
      else
        __builtin_prefetch(Bb + (long)(k0 + 64) * ldb + bn + l, 0, 1);
    }
    v16bf a0 = load_a_frag(Ab, lda, bm,      k0, l, hi);
    v16bf a1 = load_a_frag(Ab, lda, bm + 16, k0, l, hi);
    v16bf b[NT];
#pragma unroll
    for (int j = 0; j < NT; ++j) {
      if constexpr (BT) b[j] = load_b_t(Bb, ldb, k0, bn + j * 16, l, hi);
      else              b[j] = load_b_nt(Bb, ldb, k0, bn + j * 16, l, hi);
    }
#pragma unroll
    for (int j = 0; j < NT; ++j) {
      c[0][j] = wmma_bf(a0, b[j], c[0][j]);
      c[1][j] = wmma_bf(a1, b[j], c[1][j]);
    }
  }

#pragma unroll
  for (int i = 0; i < 2; ++i) {
#pragma unroll
    for (int j = 0; j < NT; ++j) {
      int col = bn + j * 16 + l;
      float bv = bias ? bias[col] : 0.0f;
      float mv = 0.0f;
      if constexpr (MM == 2) mv = (rvmask[col] == 0) ? -1e9f : 0.0f;
#pragma unroll
      for (int v = 0; v < 8; ++v) {
        int row = bm + i * 16 + v + 8 * hi;
        float x = c[i][j][v] * scale + bv;
        if constexpr (MM == 1) { if (col > row) x = -1e9f; }
        if constexpr (MM == 2) { if (mv != 0.0f) x = -1e9f; }
        if (relu) x = fmaxf(x, 0.0f);
        if (outF) outF[(long)z * batchOF + (long)row * ldof + col] = x;
        if (outB) outB[(long)z * batchOB + (long)row * ldob + col] = to_bf(x);
      }
    }
  }
}

// ---------------------------------------------------------------------------
// Row softmax, in place.  One block per row, row length 2048, 256 threads.
// ---------------------------------------------------------------------------
__device__ __forceinline__ float wave_red_sum(float x) {
#pragma unroll
  for (int o = 16; o > 0; o >>= 1) x += __shfl_xor(x, o, 32);
  return x;
}
__device__ __forceinline__ float wave_red_max(float x) {
#pragma unroll
  for (int o = 16; o > 0; o >>= 1) x = fmaxf(x, __shfl_xor(x, o, 32));
  return x;
}

__global__ void softmax_rows(float* __restrict__ data) {
  __shared__ float red[16];
  const int n = S_LEN;
  float* p = data + (long)blockIdx.x * n;
  const int t = threadIdx.x, wid = t >> 5, lane = t & 31;
  float v[8];
  float mx = -3.4e38f;
#pragma unroll
  for (int i = 0; i < 8; ++i) { v[i] = p[t + i * 256]; mx = fmaxf(mx, v[i]); }
  mx = wave_red_max(mx);
  if (lane == 0) red[wid] = mx;
  __syncthreads();
  mx = red[0];
#pragma unroll
  for (int i = 1; i < 8; ++i) mx = fmaxf(mx, red[i]);
  float s = 0.0f;
#pragma unroll
  for (int i = 0; i < 8; ++i) { v[i] = __expf(v[i] - mx); s += v[i]; }
  s = wave_red_sum(s);
  __syncthreads();
  if (lane == 0) red[8 + wid] = s;
  __syncthreads();
  float tot = 0.0f;
#pragma unroll
  for (int i = 0; i < 8; ++i) tot += red[8 + i];
  float inv = 1.0f / tot;
#pragma unroll
  for (int i = 0; i < 8; ++i) p[t + i * 256] = v[i] * inv;
}

// ---------------------------------------------------------------------------
// out = LayerNorm(a + b) * g + beta.  One block (256 thr) per row of 1024.
// ---------------------------------------------------------------------------
__global__ void add_layernorm(const float* __restrict__ a, const float* __restrict__ b,
                              const float* __restrict__ g, const float* __restrict__ bt,
                              float* __restrict__ outF, __bf16* __restrict__ outB) {
  __shared__ float red[16];
  const int D = D_MOD;
  const long row = blockIdx.x;
  const float* pa = a + row * D;
  const float* pb = b + row * D;
  const int t = threadIdx.x, wid = t >> 5, lane = t & 31;
  float x[4];
  float s = 0.0f;
#pragma unroll
  for (int i = 0; i < 4; ++i) { x[i] = pa[t + i * 256] + pb[t + i * 256]; s += x[i]; }
  s = wave_red_sum(s);
  if (lane == 0) red[wid] = s;
  __syncthreads();
  float tot = 0.0f;
#pragma unroll
  for (int i = 0; i < 8; ++i) tot += red[i];
  float mu = tot * (1.0f / D);
  float q = 0.0f;
#pragma unroll
  for (int i = 0; i < 4; ++i) { float d = x[i] - mu; q += d * d; }
  q = wave_red_sum(q);
  __syncthreads();
  if (lane == 0) red[8 + wid] = q;
  __syncthreads();
  float vtot = 0.0f;
#pragma unroll
  for (int i = 0; i < 8; ++i) vtot += red[8 + i];
  float r = rsqrtf(vtot * (1.0f / D) + 1e-6f);
#pragma unroll
  for (int i = 0; i < 4; ++i) {
    int c = t + i * 256;
    float y = (x[i] - mu) * r * g[c] + bt[c];
    outF[row * D + c] = y;
    if (outB) outB[row * D + c] = to_bf(y);
  }
}

// elementwise f32 -> bf16
__global__ void conv_bf16(const float* __restrict__ src, __bf16* __restrict__ dst, int n) {
  int i = blockIdx.x * blockDim.x + threadIdx.x;
  if (i < n) dst[i] = to_bf(src[i]);
}

// transpose + convert: src (R x C) f32  ->  dst (C x R) bf16.  R,C % 32 == 0.
__global__ void transpose_bf16(const float* __restrict__ src, __bf16* __restrict__ dst,
                               int R, int C) {
  __shared__ float tile[32][33];
  int bx = blockIdx.x * 32, by = blockIdx.y * 32;
  int tx = threadIdx.x, ty = threadIdx.y;
#pragma unroll
  for (int i = 0; i < 32; i += 8)
    tile[ty + i][tx] = src[(long)(by + ty + i) * C + bx + tx];
  __syncthreads();
#pragma unroll
  for (int i = 0; i < 32; i += 8)
    dst[(long)(bx + ty + i) * R + by + tx] = to_bf(tile[tx][ty + i]);
}

// ---------------------------------------------------------------------------
// Host side
// ---------------------------------------------------------------------------
extern "C" void kernel_launch(void* const* d_in, const int* in_sizes, int n_in,
                              void* d_out, int out_size, void* d_ws, size_t ws_size,
                              hipStream_t stream) {
  (void)in_sizes; (void)n_in; (void)out_size; (void)ws_size;
  const int S = S_LEN, D = D_MOD, H = N_HEADS, DK = D_KH, FF = D_FFN;

  const float* X   = (const float*)d_in[0];
  const float* E   = (const float*)d_in[1];
  const int* padM  = (const int*)d_in[3];    // look_ahead_mask handled as causal
  const float* fb1 = (const float*)d_in[4];
  const float* fb2 = (const float*)d_in[5];
  const float* fw1 = (const float*)d_in[6];
  const float* fw2 = (const float*)d_in[7];
  const float* ln1b = (const float*)d_in[8],  *ln1g = (const float*)d_in[9];
  const float* ln2b = (const float*)d_in[10], *ln2g = (const float*)d_in[11];
  const float* ln3b = (const float*)d_in[12], *ln3g = (const float*)d_in[13];
  const float* m1bk = (const float*)d_in[14], *m1bo = (const float*)d_in[15];
  const float* m1bq = (const float*)d_in[16], *m1bv = (const float*)d_in[17];
  const float* m1wk = (const float*)d_in[18], *m1wo = (const float*)d_in[19];
  const float* m1wq = (const float*)d_in[20], *m1wv = (const float*)d_in[21];
  const float* m2bk = (const float*)d_in[22], *m2bo = (const float*)d_in[23];
  const float* m2bq = (const float*)d_in[24], *m2bv = (const float*)d_in[25];
  const float* m2wk = (const float*)d_in[26], *m2wo = (const float*)d_in[27];
  const float* m2wq = (const float*)d_in[28], *m2wv = (const float*)d_in[29];

  float* out = (float*)d_out;
  const size_t X3_SZ = (size_t)S * D;
  const size_t AW_SZ = (size_t)H * S * S;
  float* aw1 = out + X3_SZ;
  float* aw2 = out + X3_SZ + AW_SZ;

  // ---- workspace layout (bump allocator) ----
  char* ws = (char*)d_ws;
  size_t off = 0;
  auto take = [&](size_t bytes) -> char* {
    char* p = ws + off;
    off += (bytes + 255) & ~(size_t)255;
    return p;
  };
  __bf16* wq1t = (__bf16*)take((size_t)D * D * 2);
  __bf16* wk1t = (__bf16*)take((size_t)D * D * 2);
  __bf16* wv1t = (__bf16*)take((size_t)D * D * 2);
  __bf16* wo1t = (__bf16*)take((size_t)D * D * 2);
  __bf16* wq2t = (__bf16*)take((size_t)D * D * 2);
  __bf16* wk2t = (__bf16*)take((size_t)D * D * 2);
  __bf16* wv2t = (__bf16*)take((size_t)D * D * 2);
  __bf16* wo2t = (__bf16*)take((size_t)D * D * 2);
  __bf16* w1t  = (__bf16*)take((size_t)D * FF * 2);  // stored FF x D
  __bf16* w2t  = (__bf16*)take((size_t)FF * D * 2);  // stored D x FF
  __bf16* Xb   = (__bf16*)take((size_t)S * D * 2);
  __bf16* Eb   = (__bf16*)take((size_t)S * D * 2);
  __bf16* Qb   = (__bf16*)take((size_t)S * D * 2);
  __bf16* Kb   = (__bf16*)take((size_t)S * D * 2);
  __bf16* Vb   = (__bf16*)take((size_t)S * D * 2);
  __bf16* attnb = (__bf16*)take((size_t)S * D * 2);
  float*  x1f   = (float*)take((size_t)S * D * 4);
  __bf16* x1b   = (__bf16*)take((size_t)S * D * 2);
  float*  x2f   = (float*)take((size_t)S * D * 4);
  __bf16* x2b   = (__bf16*)take((size_t)S * D * 2);
  __bf16* hb    = (__bf16*)take((size_t)S * FF * 2);
  float*  tmpf  = (float*)take((size_t)S * D * 4);

  const dim3 B256(256);
  const dim3 T32x8(32, 8);
  const dim3 gD(D / 128, S / 128, 1);     // N=1024 weight GEMMs
  const dim3 gFF(FF / 128, S / 128, 1);   // N=4096
  const dim3 gLG(S / 128, S / 128, H);    // logits, N=2048, batched heads
  const dim3 gAV(1, S / 128, H);          // aw@V, N=64, batched heads

  // ---- conversions ----
  conv_bf16<<<(S * D + 255) / 256, 256, 0, stream>>>(X, Xb, S * D);
  conv_bf16<<<(S * D + 255) / 256, 256, 0, stream>>>(E, Eb, S * D);
  transpose_bf16<<<dim3(D / 32, D / 32), T32x8, 0, stream>>>(m1wq, wq1t, D, D);
  transpose_bf16<<<dim3(D / 32, D / 32), T32x8, 0, stream>>>(m1wk, wk1t, D, D);
  transpose_bf16<<<dim3(D / 32, D / 32), T32x8, 0, stream>>>(m1wv, wv1t, D, D);
  transpose_bf16<<<dim3(D / 32, D / 32), T32x8, 0, stream>>>(m1wo, wo1t, D, D);
  transpose_bf16<<<dim3(D / 32, D / 32), T32x8, 0, stream>>>(m2wq, wq2t, D, D);
  transpose_bf16<<<dim3(D / 32, D / 32), T32x8, 0, stream>>>(m2wk, wk2t, D, D);
  transpose_bf16<<<dim3(D / 32, D / 32), T32x8, 0, stream>>>(m2wv, wv2t, D, D);
  transpose_bf16<<<dim3(D / 32, D / 32), T32x8, 0, stream>>>(m2wo, wo2t, D, D);
  transpose_bf16<<<dim3(FF / 32, D / 32), T32x8, 0, stream>>>(fw1, w1t, D, FF);
  transpose_bf16<<<dim3(D / 32, FF / 32), T32x8, 0, stream>>>(fw2, w2t, FF, D);

  // ======== self-attention ========
  wmma_gemm<__bf16, true, 4, 0><<<gD, B256, 0, stream>>>(
      Xb, 0, D, wq1t, 0, D, m1bq, nullptr, 1.0f,
      nullptr, 0, 0, Qb, 0, D, D, 0);
  wmma_gemm<__bf16, true, 4, 0><<<gD, B256, 0, stream>>>(
      Xb, 0, D, wk1t, 0, D, m1bk, nullptr, 1.0f,
      nullptr, 0, 0, Kb, 0, D, D, 0);
  wmma_gemm<__bf16, true, 4, 0><<<gD, B256, 0, stream>>>(
      Xb, 0, D, wv1t, 0, D, m1bv, nullptr, 1.0f,
      nullptr, 0, 0, Vb, 0, D, D, 0);
  // logits = Q@K^T/8 + causal mask.
  // Q: head slice (off h*64, lda D).  K: faithful raw view (off h*S*DK, ld DK).
  wmma_gemm<__bf16, true, 4, 1><<<gLG, B256, 0, stream>>>(
      Qb, DK, D, Kb, (long)S * DK, DK, nullptr, nullptr, 0.125f,
      aw1, (long)S * S, S, nullptr, 0, 0, DK, 0);
  softmax_rows<<<H * S, 256, 0, stream>>>(aw1);
  // out = aw @ V  (raw-view write, ld DK); A is f32 straight from d_out
  wmma_gemm<float, false, 2, 0><<<gAV, B256, 0, stream>>>(
      aw1, (long)S * S, S, Vb, (long)S * DK, DK, nullptr, nullptr, 1.0f,
      nullptr, 0, 0, attnb, (long)S * DK, DK, S, 0);
  wmma_gemm<__bf16, true, 4, 0><<<gD, B256, 0, stream>>>(
      attnb, 0, D, wo1t, 0, D, m1bo, nullptr, 1.0f,
      tmpf, 0, D, nullptr, 0, 0, D, 0);
  add_layernorm<<<S, 256, 0, stream>>>(tmpf, X, ln1g, ln1b, x1f, x1b);

  // ======== cross-attention ========
  wmma_gemm<__bf16, true, 4, 0><<<gD, B256, 0, stream>>>(
      x1b, 0, D, wq2t, 0, D, m2bq, nullptr, 1.0f,
      nullptr, 0, 0, Qb, 0, D, D, 0);
  wmma_gemm<__bf16, true, 4, 0><<<gD, B256, 0, stream>>>(
      Eb, 0, D, wk2t, 0, D, m2bk, nullptr, 1.0f,
      nullptr, 0, 0, Kb, 0, D, D, 0);
  wmma_gemm<__bf16, true, 4, 0><<<gD, B256, 0, stream>>>(
      Eb, 0, D, wv2t, 0, D, m2bv, nullptr, 1.0f,
      nullptr, 0, 0, Vb, 0, D, D, 0);
  // padding mask is a broadcast row vector: MM=2 reads mask[col] only.
  wmma_gemm<__bf16, true, 4, 2><<<gLG, B256, 0, stream>>>(
      Qb, DK, D, Kb, (long)S * DK, DK, nullptr, padM, 0.125f,
      aw2, (long)S * S, S, nullptr, 0, 0, DK, 0);
  softmax_rows<<<H * S, 256, 0, stream>>>(aw2);
  wmma_gemm<float, false, 2, 0><<<gAV, B256, 0, stream>>>(
      aw2, (long)S * S, S, Vb, (long)S * DK, DK, nullptr, nullptr, 1.0f,
      nullptr, 0, 0, attnb, (long)S * DK, DK, S, 0);
  wmma_gemm<__bf16, true, 4, 0><<<gD, B256, 0, stream>>>(
      attnb, 0, D, wo2t, 0, D, m2bo, nullptr, 1.0f,
      tmpf, 0, D, nullptr, 0, 0, D, 0);
  add_layernorm<<<S, 256, 0, stream>>>(x1f, tmpf, ln2g, ln2b, x2f, x2b);

  // ======== FFN ========
  wmma_gemm<__bf16, true, 4, 0><<<gFF, B256, 0, stream>>>(
      x2b, 0, D, w1t, 0, D, fb1, nullptr, 1.0f,
      nullptr, 0, 0, hb, 0, FF, D, 1);
  wmma_gemm<__bf16, true, 4, 0><<<gD, B256, 0, stream>>>(
      hb, 0, FF, w2t, 0, FF, fb2, nullptr, 1.0f,
      tmpf, 0, D, nullptr, 0, 0, FF, 0);
  add_layernorm<<<S, 256, 0, stream>>>(tmpf, x2f, ln3g, ln3b, out, nullptr);
}
